// DynamicEuclideanCodebook_6382321402116
// MI455X (gfx1250) — compile-verified, matrix operand order checked by reading, not run on hardware
//
#include <hip/hip_runtime.h>
#include <hip/hip_bf16.h>

typedef __attribute__((ext_vector_type(16))) __bf16 v16bf;
typedef __attribute__((ext_vector_type(8)))  float  v8f;

#define NDIM    256
#define CB      4
#define HD      64
#define KCODES  256
#define CTXD    256
#define NNODES  4096
#define W2COLS  (CB * KCODES * HD)   // 262144

// ---------------------------------------------------------------------------
// Kernel 1: h = leaky_relu([context | type_table[node_type]] @ W1 + b1) -> bf16
// Tiny GEMM (0.8 GFLOP); plain fp32 VALU, coalesced over output columns.
// ---------------------------------------------------------------------------
__global__ void k_compute_h(const float* __restrict__ context,
                            const int*   __restrict__ node_type,
                            const float* __restrict__ type_table,
                            const float* __restrict__ W1,
                            const float* __restrict__ b1,
                            __hip_bfloat16* __restrict__ h_bf) {
  const int n = blockIdx.x;       // node
  const int j = threadIdx.x;      // output dim, 256 threads
  const float* ctx = context + (size_t)n * CTXD;
  const float* te  = type_table + (size_t)node_type[n] * (NDIM / 2);
  float acc = b1[j];
  for (int d = 0; d < CTXD; ++d)
    acc = fmaf(ctx[d], W1[(size_t)d * NDIM + j], acc);
  for (int d = 0; d < NDIM / 2; ++d)
    acc = fmaf(te[d], W1[(size_t)(CTXD + d) * NDIM + j], acc);
  acc = acc > 0.0f ? acc : 0.01f * acc;   // torch leaky_relu default slope
  h_bf[(size_t)n * NDIM + j] = __float2bfloat16(acc);
}

// ---------------------------------------------------------------------------
// Kernel 2: W2 [256 x 262144] f32 row-major -> W2t [262144 x 256] bf16.
// 128 MB result fits in the 192 MB L2, so the main kernel streams it from L2.
// ---------------------------------------------------------------------------
__global__ void k_transpose_w2(const float* __restrict__ W2,
                               __hip_bfloat16* __restrict__ W2t) {
  const size_t col = (size_t)blockIdx.x * blockDim.x + threadIdx.x;
  __hip_bfloat16* dst = W2t + col * NDIM;
  for (int d = 0; d < NDIM; ++d)
    dst[d] = __float2bfloat16(W2[(size_t)d * W2COLS + col]);
}

// ---------------------------------------------------------------------------
// Kernel 3: fused  delta-GEMM + eff + distance + argmin + gather.
// One wave32 per block; block = (node_tile of 16, codebook c).
// Per code k: 4 column tiles x 8 K-steps = 32 x v_wmma_f32_16x16x32_bf16.
// ---------------------------------------------------------------------------
__global__ __launch_bounds__(32)
void k_vq(const float* __restrict__ x,
          const __hip_bfloat16* __restrict__ h_bf,
          const __hip_bfloat16* __restrict__ W2t,
          const float* __restrict__ embed,
          const float* __restrict__ b2,
          float* __restrict__ out) {
  const int tile   = blockIdx.x;          // 0..255 node tiles
  const int c      = blockIdx.y;          // 0..3 codebooks
  const int lane   = threadIdx.x;         // 0..31
  const int laneN  = lane & 15;           // column within 16-wide tile (D layout)
  const int laneHi = lane >> 4;           // half-wave -> node row +8
  const int node0  = tile * 16;

  // A fragments: h_bf tile [16 nodes x 256 dims], 8 K-steps, kept in VGPRs
  // (reused for all 256 codes). Per-lane 32B contiguous loads.
  v16bf a[8];
  {
    const __hip_bfloat16* ap =
        h_bf + (size_t)(node0 + laneN) * NDIM + laneHi * 16;
#pragma unroll
    for (int s = 0; s < 8; ++s)
      a[s] = *(const v16bf*)(const void*)(ap + s * 32);
  }

  // x preloaded in exactly the WMMA D layout: lane=column h, vgpr v=node row.
  float xr[4][8];
#pragma unroll
  for (int t = 0; t < 4; ++t)
#pragma unroll
    for (int v = 0; v < 8; ++v)
      xr[t][v] = x[(size_t)(node0 + v + laneHi * 8) * NDIM + c * HD + t * 16 + laneN];

  // |x|^2 per node via 16-lane butterfly (all lanes end with the full sum).
  float xsq[8];
#pragma unroll
  for (int v = 0; v < 8; ++v) {
    float s = 0.0f;
#pragma unroll
    for (int t = 0; t < 4; ++t) s = fmaf(xr[t][v], xr[t][v], s);
#pragma unroll
    for (int m = 8; m >= 1; m >>= 1) s += __shfl_xor(s, m, 16);
    xsq[v] = s;
  }

  float best_d[8], best_eff[4][8];
  int   best_k[8];
#pragma unroll
  for (int v = 0; v < 8; ++v) {
    best_d[v] = 3.4e38f;
    best_k[v] = 0;
#pragma unroll
    for (int t = 0; t < 4; ++t) best_eff[t][v] = 0.0f;
  }

  const size_t colbase0 = (size_t)c * KCODES * HD;   // c * 16384
  const v8f vzero = {};

  for (int k = 0; k < KCODES; ++k) {
    const size_t col0 = colbase0 + (size_t)k * HD;

    // Codebook base + bias for this code, one scalar per lane-column.
    float ev[4], bv[4];
#pragma unroll
    for (int t = 0; t < 4; ++t) {
      ev[t] = embed[col0 + t * 16 + laneN];
      bv[t] = b2[col0 + t * 16 + laneN];
    }

    v8f acc[4];
#pragma unroll
    for (int t = 0; t < 4; ++t) acc[t] = vzero;

    // delta block [16 nodes x 64 h] = h @ W2[:, code k columns]
#pragma unroll
    for (int s = 0; s < 8; ++s) {
#pragma unroll
      for (int t = 0; t < 4; ++t) {
        const __hip_bfloat16* bp =
            W2t + (col0 + t * 16 + laneN) * NDIM + s * 32 + laneHi * 16;
        v16bf bf = *(const v16bf*)(const void*)(bp);
        acc[t] = __builtin_amdgcn_wmma_f32_16x16x32_bf16(
            false, a[s], false, bf, (short)0, acc[t], false, false);
      }
    }

    // Prefetch next code's W2t block into cache (global_prefetch_b8).
    if (k + 1 < KCODES) {
#pragma unroll
      for (int t = 0; t < 4; ++t)
        __builtin_prefetch(
            (const void*)(W2t + (col0 + HD + t * 16 + laneN) * NDIM + laneHi * 128),
            0, 0);
    }

    // eff = delta + embed + b2 ; accumulate x.eff and |eff|^2 per node.
    float xe[8], esq[8];
#pragma unroll
    for (int v = 0; v < 8; ++v) { xe[v] = 0.0f; esq[v] = 0.0f; }
#pragma unroll
    for (int t = 0; t < 4; ++t)
#pragma unroll
      for (int v = 0; v < 8; ++v) {
        float eff = acc[t][v] + ev[t] + bv[t];
        xe[v]  = fmaf(xr[t][v], eff, xe[v]);
        esq[v] = fmaf(eff, eff, esq[v]);
      }

#pragma unroll
    for (int v = 0; v < 8; ++v) {
#pragma unroll
      for (int m = 8; m >= 1; m >>= 1) {
        xe[v]  += __shfl_xor(xe[v],  m, 16);
        esq[v] += __shfl_xor(esq[v], m, 16);
      }
      float dist = xsq[v] - 2.0f * xe[v] + esq[v];
      bool better = dist < best_d[v];             // uniform within 16-lane group
      best_d[v] = better ? dist : best_d[v];
      best_k[v] = better ? k : best_k[v];
#pragma unroll
      for (int t = 0; t < 4; ++t) {
        float eff = acc[t][v] + ev[t] + bv[t];    // rematerialize, saves 32 VGPRs
        best_eff[t][v] = better ? eff : best_eff[t][v];
      }
    }
  }

  // quantized[n, c*64 + h] — best_eff is already in D-layout.
#pragma unroll
  for (int t = 0; t < 4; ++t)
#pragma unroll
    for (int v = 0; v < 8; ++v)
      out[(size_t)(node0 + v + laneHi * 8) * NDIM + c * HD + t * 16 + laneN] =
          best_eff[t][v];

  // embed_ind[n, c] (int32) concatenated after quantized.
  if (laneN == 0) {
    int* idx = (int*)(out + (size_t)NNODES * NDIM);
#pragma unroll
    for (int v = 0; v < 8; ++v)
      idx[(node0 + v + laneHi * 8) * CB + c] = best_k[v];
  }
}

// ---------------------------------------------------------------------------
extern "C" void kernel_launch(void* const* d_in, const int* in_sizes, int n_in,
                              void* d_out, int out_size, void* d_ws, size_t ws_size,
                              hipStream_t stream) {
  const float* x          = (const float*)d_in[0];
  const int*   node_type  = (const int*)  d_in[1];
  const float* context    = (const float*)d_in[2];
  const float* embed      = (const float*)d_in[3];
  const float* type_table = (const float*)d_in[4];
  const float* W1         = (const float*)d_in[5];
  const float* b1         = (const float*)d_in[6];
  const float* W2         = (const float*)d_in[7];
  const float* b2         = (const float*)d_in[8];
  float* out = (float*)d_out;

  // Workspace layout: h_bf (2 MB) | W2t bf16 (128 MB)
  __hip_bfloat16* h_bf = (__hip_bfloat16*)d_ws;
  __hip_bfloat16* W2t  =
      (__hip_bfloat16*)((char*)d_ws + (size_t)NNODES * NDIM * sizeof(__hip_bfloat16));

  k_compute_h<<<NNODES, NDIM, 0, stream>>>(context, node_type, type_table, W1, b1, h_bf);
  k_transpose_w2<<<W2COLS / 256, 256, 0, stream>>>(W2, W2t);

  dim3 grid(NNODES / 16, CB);
  k_vq<<<grid, 32, 0, stream>>>(x, h_bf, W2t, embed, b2, out);
}
